// Decoder_26130581029610
// MI455X (gfx1250) — matrix operand order repaired and checked by reading, not compiled, not run on hardware
//
#include <hip/hip_runtime.h>
#include <cstdint>
#include <cstddef>

// ---------------- types ----------------
typedef unsigned short ushort_t;
typedef __bf16 bf16_t;
typedef bf16_t v16bf __attribute__((ext_vector_type(16)));
typedef float  v8f   __attribute__((ext_vector_type(8)));

union Frag16 {
  v16bf   bf;
  ushort_t u[16];
};

__device__ __forceinline__ ushort_t f2bf(float x) {
  unsigned u = __float_as_uint(x);
  u += 0x7FFFu + ((u >> 16) & 1u);   // round-to-nearest-even
  return (ushort_t)(u >> 16);
}

__device__ __forceinline__ v8f wmma_bf16(const Frag16& a, const Frag16& b, v8f c) {
  return __builtin_amdgcn_wmma_f32_16x16x32_bf16(false, a.bf, false, b.bf, (short)0, c, false, false);
}

// load A fragment (row-major bf16 [.,K]); lane<16: K=+0..7,+16..23 ; lane>=16: K=+8..15,+24..31
__device__ __forceinline__ void loadA(Frag16& f, const ushort_t* base, int row, int ldk, int k0, int l16, bool hi) {
  const ushort_t* p = base + (size_t)row * ldk + k0 + (hi ? 8 : 0);
  *(uint4*)&f.u[0] = *(const uint4*)p;
  *(uint4*)&f.u[8] = *(const uint4*)(p + 16);
}
// load B fragment from col-major storage (BT row-major [N,K]); lane holds col n; lane<16: K=+0..7,+16..23 ; lane>=16: K=+8..15,+24..31
__device__ __forceinline__ void loadB(Frag16& f, const ushort_t* base, int col, int ldk, int k0, bool hi) {
  const ushort_t* p = base + (size_t)col * ldk + k0 + (hi ? 8 : 0);
  *(uint4*)&f.u[0] = *(const uint4*)p;
  *(uint4*)&f.u[8] = *(const uint4*)(p + 16);
}

// ---------------- constants ----------------
#define DD   512
#define HH   8
#define DKK  64
#define LL   6
#define VV   32000
#define BB   16
#define SS   128
#define MM   196
#define TT   (BB*SS)      // 2048
#define MT   (BB*MM)      // 3136
#define BHH  (BB*HH)      // 128
#define SKP_CA 224

// epilogue modes
#define MODE_F32        0
#define MODE_BF16_RELU  2
#define MODE_HEAD       3   // bf16 -> [(b*H+h)*cap + s]*64 + dk
#define MODE_VT         5   // bf16 -> [(b*H+h)*64 + dk]*cap + s

// ---------------- generic WMMA GEMM ----------------
// C[M,N] = A[M,K](bf16 rowmajor) * B (stored as BT[N,K] bf16) + bias, epilogue by mode.
// block 256 = 8 waves (2x4 grid), wave tile = 32x64 (2 A-frags x 4 B-frags -> 8 WMMA / 6 frag loads).
// WG covers 64x256. Requires M%64==0, N%256==0, K%32==0.
__global__ void gemm_wmma(const ushort_t* __restrict__ A, const ushort_t* __restrict__ BT,
                          const float* __restrict__ bias, void* __restrict__ out,
                          int Mm, int Nn, int Kk, int mode, int seq, int cap)
{
  int lane = threadIdx.x & 31;
  int wid  = threadIdx.x >> 5;
  int wm = wid >> 2, wn = wid & 3;
  int mBase = blockIdx.x * 64 + wm * 32;
  int nBase = blockIdx.y * 256 + wn * 64;
  bool hi = lane >= 16;
  int l16 = lane & 15;

  v8f acc[2][4] = {};
  for (int k0 = 0; k0 < Kk; k0 += 32) {
    Frag16 a[2], b[4];
#pragma unroll
    for (int t = 0; t < 2; t++) loadA(a[t], A, mBase + t * 16 + l16, Kk, k0, l16, hi);
#pragma unroll
    for (int t = 0; t < 4; t++) loadB(b[t], BT, nBase + t * 16 + l16, Kk, k0, hi);
#pragma unroll
    for (int i = 0; i < 2; i++)
#pragma unroll
      for (int j = 0; j < 4; j++)
        acc[i][j] = wmma_bf16(a[i], b[j], acc[i][j]);
  }

#pragma unroll
  for (int j = 0; j < 4; j++) {
    int col = nBase + j * 16 + l16;
    float bv = bias[col];
#pragma unroll
    for (int i = 0; i < 2; i++) {
      int rowB = mBase + i * 16 + (hi ? 8 : 0);
#pragma unroll
      for (int r = 0; r < 8; r++) {
        int row = rowB + r;
        float v = acc[i][j][r] + bv;
        if (mode == MODE_F32) {
          ((float*)out)[(size_t)row * Nn + col] = v;
        } else if (mode == MODE_BF16_RELU) {
          ((ushort_t*)out)[(size_t)row * Nn + col] = f2bf(v > 0.f ? v : 0.f);
        } else if (mode == MODE_HEAD) {
          int b_ = row / seq, s_ = row - b_ * seq;
          int h_ = col >> 6, dk = col & 63;
          ((ushort_t*)out)[((size_t)(b_ * HH + h_) * cap + s_) * 64 + dk] = f2bf(v);
        } else { // MODE_VT
          int b_ = row / seq, s_ = row - b_ * seq;
          int h_ = col >> 6, dk = col & 63;
          ((ushort_t*)out)[((size_t)(b_ * HH + h_) * 64 + dk) * cap + s_] = f2bf(v);
        }
      }
    }
  }
}

// ---------------- attention: scores = Q K^T / 64 ----------------
// Q: [BH, SS, 64] bf16 ; Kb: [BH, SKP, 64] bf16 ; scores: [BH, SS, SKP] f32
__global__ void attn_scores(const ushort_t* __restrict__ Q, const ushort_t* __restrict__ Kb,
                            float* __restrict__ scores, int SKP, float scale)
{
  int bh = blockIdx.x;
  int tiles_n = SKP >> 4;
  int total = (SS >> 4) * tiles_n;
  int tile = blockIdx.y * 8 + (threadIdx.x >> 5);
  if (tile >= total) return;
  int tm = tile / tiles_n, tn = tile - tm * tiles_n;
  int lane = threadIdx.x & 31, l16 = lane & 15;
  bool hi = lane >= 16;
  const ushort_t* q  = Q  + (size_t)bh * SS * 64;
  const ushort_t* kb = Kb + (size_t)bh * SKP * 64;
  v8f acc = {};
#pragma unroll
  for (int k0 = 0; k0 < 64; k0 += 32) {
    Frag16 a, b;
    loadA(a, q,  tm * 16 + l16, 64, k0, l16, hi);
    loadB(b, kb, tn * 16 + l16, 64, k0, hi);
    acc = wmma_bf16(a, b, acc);
  }
#pragma unroll
  for (int r = 0; r < 8; r++) {
    int row = tm * 16 + r + (hi ? 8 : 0);
    int col = tn * 16 + l16;
    scores[((size_t)bh * SS + row) * SKP + col] = acc[r] * scale;
  }
}

// ---------------- softmax (one wave per row) ----------------
__global__ void softmax_rows(const float* __restrict__ scores, ushort_t* __restrict__ attn,
                             int SKP, int valid_fixed, int causal)
{
  int row = blockIdx.x * 8 + (threadIdx.x >> 5);   // BH*SS rows
  int lane = threadIdx.x & 31;
  int q = row & (SS - 1);
  int valid = causal ? (q + 1) : valid_fixed;
  const float* src = scores + (size_t)row * SKP;
  float mx = -3.4e38f;
  for (int j = lane; j < valid; j += 32) mx = fmaxf(mx, src[j]);
#pragma unroll
  for (int o = 16; o >= 1; o >>= 1) mx = fmaxf(mx, __shfl_xor(mx, o));
  float sum = 0.f;
  for (int j = lane; j < valid; j += 32) sum += __expf(src[j] - mx);
#pragma unroll
  for (int o = 16; o >= 1; o >>= 1) sum += __shfl_xor(sum, o);
  float inv = 1.0f / sum;
  ushort_t* dst = attn + (size_t)row * SKP;
  for (int j = lane; j < SKP; j += 32)
    dst[j] = (j < valid) ? f2bf(__expf(src[j] - mx) * inv) : (ushort_t)0;
}

// ---------------- O = attn @ V  (V stored transposed per head) ----------------
// attn: [BH, SS, SKP] bf16 ; vT: [BH, 64, SKP] bf16 ; O: bf16 [(b*SS+s)*512 + h*64+dk]
__global__ void attn_value(const ushort_t* __restrict__ attn, const ushort_t* __restrict__ vT,
                           ushort_t* __restrict__ O, int SKP)
{
  int bh = blockIdx.x;
  int b_ = bh >> 3, h_ = bh & 7;
  int tile = blockIdx.y * 8 + (threadIdx.x >> 5);   // 32 tiles: 8 (q) x 4 (dk)
  int tm = tile >> 2, tn = tile & 3;
  int lane = threadIdx.x & 31, l16 = lane & 15;
  bool hi = lane >= 16;
  const ushort_t* a_ = attn + (size_t)bh * SS * SKP;
  const ushort_t* v_ = vT + (size_t)bh * 64 * SKP;
  v8f acc = {};
  for (int k0 = 0; k0 < SKP; k0 += 32) {
    Frag16 a, b;
    loadA(a, a_, tm * 16 + l16, SKP, k0, l16, hi);
    loadB(b, v_, tn * 16 + l16, SKP, k0, hi);
    acc = wmma_bf16(a, b, acc);
  }
#pragma unroll
  for (int r = 0; r < 8; r++) {
    int s_ = tm * 16 + r + (hi ? 8 : 0);
    int dk = tn * 16 + l16;
    O[((size_t)(b_ * SS + s_)) * DD + h_ * 64 + dk] = f2bf(acc[r]);
  }
}

// ---------------- residual + layernorm (block per row of 512) ----------------
__global__ void add_ln(const float* __restrict__ x, const float* __restrict__ resid,
                       const float* __restrict__ g, const float* __restrict__ bta,
                       float* __restrict__ hout, ushort_t* __restrict__ hbf, float eps)
{
  int row = blockIdx.x;
  int t = threadIdx.x;
  __shared__ float red[256];
  size_t base = (size_t)row * DD;
  float v0 = x[base + t] + resid[base + t];
  float v1 = x[base + 256 + t] + resid[base + 256 + t];
  red[t] = v0 + v1; __syncthreads();
  for (int s = 128; s > 0; s >>= 1) { if (t < s) red[t] += red[t + s]; __syncthreads(); }
  float mu = red[0] * (1.f / 512.f); __syncthreads();
  float d0 = v0 - mu, d1 = v1 - mu;
  red[t] = d0 * d0 + d1 * d1; __syncthreads();
  for (int s = 128; s > 0; s >>= 1) { if (t < s) red[t] += red[t + s]; __syncthreads(); }
  float inv = rsqrtf(red[0] * (1.f / 512.f) + eps);
  float y0 = d0 * inv * g[t] + bta[t];
  float y1 = d1 * inv * g[256 + t] + bta[256 + t];
  hout[base + t] = y0; hout[base + 256 + t] = y1;
  hbf[base + t] = f2bf(y0); hbf[base + 256 + t] = f2bf(y1);
}

// ---------------- embedding ----------------
__global__ void embed_kernel(const int* __restrict__ x, const float* __restrict__ emb,
                             const float* __restrict__ pos, float* __restrict__ h,
                             ushort_t* __restrict__ hbf)
{
  int row = blockIdx.x;          // b*S+s
  int s = row & (SS - 1);
  int tok = x[row];
  for (int d = threadIdx.x; d < DD; d += 256) {
    float v = emb[(size_t)tok * DD + d] + pos[(size_t)s * DD + d];
    h[(size_t)row * DD + d] = v;
    hbf[(size_t)row * DD + d] = f2bf(v);
  }
}

// ---------------- transpose-convert f32[K,N] -> bf16[N,K] ----------------
__global__ void transpose_f32_bf16(const float* __restrict__ in, ushort_t* __restrict__ out,
                                   int Kk, int Nn)
{
  __shared__ float tile[32][33];
  int nb = blockIdx.x * 32, kb = blockIdx.y * 32;
  int tx = threadIdx.x & 31, ty = threadIdx.x >> 5;
  for (int r = ty; r < 32; r += 8)
    tile[r][tx] = in[(size_t)(kb + r) * Nn + nb + tx];
  __syncthreads();
  for (int r = ty; r < 32; r += 8)
    out[(size_t)(nb + r) * Kk + kb + tx] = f2bf(tile[tx][r]);
}

__global__ void conv_bf16(const float* __restrict__ in, ushort_t* __restrict__ out, int n)
{
  int i = blockIdx.x * 256 + threadIdx.x;
  if (i < n) out[i] = f2bf(in[i]);
}

__global__ void zero_u16(ushort_t* __restrict__ p, int n)
{
  int i = blockIdx.x * 256 + threadIdx.x;
  if (i < n) p[i] = 0;
}

// ---------------- host driver ----------------
extern "C" void kernel_launch(void* const* d_in, const int* in_sizes, int n_in,
                              void* d_out, int out_size, void* d_ws, size_t ws_size,
                              hipStream_t stream)
{
  (void)in_sizes; (void)n_in; (void)out_size; (void)ws_size;
  // inputs in setup_inputs() dict order
  const int*   x      = (const int*)  d_in[0];
  const float* memory = (const float*)d_in[1];
  const float* embw   = (const float*)d_in[3];
  const float* pos    = (const float*)d_in[4];
  const float* sa_qw = (const float*)d_in[5],  *sa_kw = (const float*)d_in[6];
  const float* sa_vw = (const float*)d_in[7],  *sa_ow = (const float*)d_in[8];
  const float* sa_qb = (const float*)d_in[9],  *sa_kb = (const float*)d_in[10];
  const float* sa_vb = (const float*)d_in[11], *sa_ob = (const float*)d_in[12];
  const float* sa_lnb = (const float*)d_in[13], *sa_lng = (const float*)d_in[14];
  const float* ca_qw = (const float*)d_in[15], *ca_kw = (const float*)d_in[16];
  const float* ca_vw = (const float*)d_in[17], *ca_ow = (const float*)d_in[18];
  const float* ca_qb = (const float*)d_in[19], *ca_kb = (const float*)d_in[20];
  const float* ca_vb = (const float*)d_in[21], *ca_ob = (const float*)d_in[22];
  const float* ca_lnb = (const float*)d_in[23], *ca_lng = (const float*)d_in[24];
  const float* ff_w1 = (const float*)d_in[25], *ff_b1 = (const float*)d_in[26];
  const float* ff_w2 = (const float*)d_in[27], *ff_b2 = (const float*)d_in[28];
  const float* ff_lng = (const float*)d_in[29], *ff_lnb = (const float*)d_in[30];
  const float* logit_w = (const float*)d_in[31], *logit_b = (const float*)d_in[32];

  // workspace carve-up
  char* ws = (char*)d_ws;
  size_t off = 0;
  auto alloc = [&](size_t bytes) -> char* {
    char* p = ws + off; off += (bytes + 255) & ~(size_t)255; return p;
  };
  float*    hf   = (float*)   alloc((size_t)TT * DD * 4);
  ushort_t* hb   = (ushort_t*)alloc((size_t)TT * DD * 2);
  ushort_t* memb = (ushort_t*)alloc((size_t)MT * DD * 2);
  ushort_t* wTq  = (ushort_t*)alloc((size_t)DD * DD * 2);
  ushort_t* wTk  = (ushort_t*)alloc((size_t)DD * DD * 2);
  ushort_t* wTv  = (ushort_t*)alloc((size_t)DD * DD * 2);
  ushort_t* wTo  = (ushort_t*)alloc((size_t)DD * DD * 2);
  ushort_t* wT1  = (ushort_t*)alloc((size_t)DD * 4 * DD * 2);
  ushort_t* wT2  = (ushort_t*)alloc((size_t)DD * 4 * DD * 2);
  ushort_t* lwT  = (ushort_t*)alloc((size_t)VV * DD * 2);
  ushort_t* qbuf = (ushort_t*)alloc((size_t)BHH * SS * 64 * 2);
  ushort_t* kbuf = (ushort_t*)alloc((size_t)BHH * SKP_CA * 64 * 2);
  ushort_t* vtbf = (ushort_t*)alloc((size_t)BHH * 64 * SKP_CA * 2);
  float*    scb  = (float*)   alloc((size_t)BHH * SS * SKP_CA * 4);
  ushort_t* atb  = (ushort_t*)alloc((size_t)BHH * SS * SKP_CA * 2);
  ushort_t* obuf = (ushort_t*)alloc((size_t)TT * DD * 2);
  float*    gout = (float*)   alloc((size_t)TT * DD * 4);
  ushort_t* ffa  = (ushort_t*)alloc((size_t)TT * 4 * DD * 2);

  const dim3 blk(256);
  const float scale = 1.0f / 64.0f;   // reference divides by d_k, not sqrt

  // prep: memory->bf16, embedding, logit_w transpose
  conv_bf16<<<(MT * DD + 255) / 256, blk, 0, stream>>>(memory, memb, MT * DD);
  embed_kernel<<<TT, blk, 0, stream>>>(x, embw, pos, hf, hb);
  transpose_f32_bf16<<<dim3(VV / 32, DD / 32), blk, 0, stream>>>(logit_w, lwT, DD, VV);

  for (int i = 0; i < LL; i++) {
    size_t wOff = (size_t)i * DD * DD, bOff = (size_t)i * DD;

    // ===== masked self-attention =====
    transpose_f32_bf16<<<dim3(16, 16), blk, 0, stream>>>(sa_qw + wOff, wTq, DD, DD);
    transpose_f32_bf16<<<dim3(16, 16), blk, 0, stream>>>(sa_kw + wOff, wTk, DD, DD);
    transpose_f32_bf16<<<dim3(16, 16), blk, 0, stream>>>(sa_vw + wOff, wTv, DD, DD);
    transpose_f32_bf16<<<dim3(16, 16), blk, 0, stream>>>(sa_ow + wOff, wTo, DD, DD);
    gemm_wmma<<<dim3(TT / 64, DD / 256), blk, 0, stream>>>(hb, wTq, sa_qb + bOff, qbuf, TT, DD, DD, MODE_HEAD, SS, SS);
    gemm_wmma<<<dim3(TT / 64, DD / 256), blk, 0, stream>>>(hb, wTk, sa_kb + bOff, kbuf, TT, DD, DD, MODE_HEAD, SS, SS);
    gemm_wmma<<<dim3(TT / 64, DD / 256), blk, 0, stream>>>(hb, wTv, sa_vb + bOff, vtbf, TT, DD, DD, MODE_VT,   SS, SS);
    attn_scores<<<dim3(BHH, 8), blk, 0, stream>>>(qbuf, kbuf, scb, SS, scale);
    softmax_rows<<<(BHH * SS) / 8, blk, 0, stream>>>(scb, atb, SS, SS, 1);
    attn_value<<<dim3(BHH, 4), blk, 0, stream>>>(atb, vtbf, obuf, SS);
    gemm_wmma<<<dim3(TT / 64, DD / 256), blk, 0, stream>>>(obuf, wTo, sa_ob + bOff, gout, TT, DD, DD, MODE_F32, 0, 0);
    add_ln<<<TT, blk, 0, stream>>>(gout, hf, sa_lng + bOff, sa_lnb + bOff, hf, hb, 1e-8f);

    // ===== cross-attention over memory =====
    transpose_f32_bf16<<<dim3(16, 16), blk, 0, stream>>>(ca_qw + wOff, wTq, DD, DD);
    transpose_f32_bf16<<<dim3(16, 16), blk, 0, stream>>>(ca_kw + wOff, wTk, DD, DD);
    transpose_f32_bf16<<<dim3(16, 16), blk, 0, stream>>>(ca_vw + wOff, wTv, DD, DD);
    transpose_f32_bf16<<<dim3(16, 16), blk, 0, stream>>>(ca_ow + wOff, wTo, DD, DD);
    zero_u16<<<(BHH * SKP_CA * 64 + 255) / 256, blk, 0, stream>>>(kbuf, BHH * SKP_CA * 64);
    zero_u16<<<(BHH * SKP_CA * 64 + 255) / 256, blk, 0, stream>>>(vtbf, BHH * SKP_CA * 64);
    gemm_wmma<<<dim3(TT / 64, DD / 256), blk, 0, stream>>>(hb,   wTq, ca_qb + bOff, qbuf, TT, DD, DD, MODE_HEAD, SS, SS);
    gemm_wmma<<<dim3(MT / 64, DD / 256), blk, 0, stream>>>(memb, wTk, ca_kb + bOff, kbuf, MT, DD, DD, MODE_HEAD, MM, SKP_CA);
    gemm_wmma<<<dim3(MT / 64, DD / 256), blk, 0, stream>>>(memb, wTv, ca_vb + bOff, vtbf, MT, DD, DD, MODE_VT,   MM, SKP_CA);
    attn_scores<<<dim3(BHH, 14), blk, 0, stream>>>(qbuf, kbuf, scb, SKP_CA, scale);
    softmax_rows<<<(BHH * SS) / 8, blk, 0, stream>>>(scb, atb, SKP_CA, MM, 0);
    attn_value<<<dim3(BHH, 4), blk, 0, stream>>>(atb, vtbf, obuf, SKP_CA);
    gemm_wmma<<<dim3(TT / 64, DD / 256), blk, 0, stream>>>(obuf, wTo, ca_ob + bOff, gout, TT, DD, DD, MODE_F32, 0, 0);
    add_ln<<<TT, blk, 0, stream>>>(gout, hf, ca_lng + bOff, ca_lnb + bOff, hf, hb, 1e-8f);

    // ===== feed-forward =====
    transpose_f32_bf16<<<dim3(64, 16), blk, 0, stream>>>(ff_w1 + (size_t)i * DD * 4 * DD, wT1, DD, 4 * DD);
    transpose_f32_bf16<<<dim3(16, 64), blk, 0, stream>>>(ff_w2 + (size_t)i * 4 * DD * DD, wT2, 4 * DD, DD);
    gemm_wmma<<<dim3(TT / 64, (4 * DD) / 256), blk, 0, stream>>>(hb, wT1, ff_b1 + (size_t)i * 4 * DD, ffa, TT, 4 * DD, DD, MODE_BF16_RELU, 0, 0);
    gemm_wmma<<<dim3(TT / 64, DD / 256), blk, 0, stream>>>(ffa, wT2, ff_b2 + bOff, gout, TT, DD, 4 * DD, MODE_F32, 0, 0);
    add_ln<<<TT, blk, 0, stream>>>(gout, hf, ff_lng + bOff, ff_lnb + bOff, hf, hb, 1e-6f);
  }

  // ===== logits =====
  gemm_wmma<<<dim3(TT / 64, VV / 256), blk, 0, stream>>>(hb, lwT, logit_b, (float*)d_out, TT, VV, DD, MODE_F32, 0, 0);
}
